// LanguageModel_43843026158036
// MI455X (gfx1250) — matrix operand order repaired and checked by reading, not compiled, not run on hardware
//
#include <hip/hip_runtime.h>

// ---------------------------------------------------------------------------
// Transformer LM forward for MI455X (gfx1250, wave32, WMMA).
// Precision: bf16 inputs to V_WMMA_F32_16X16X32_BF16, f32 accumulate,
// fp32 residual master. Compute-bound => maximize K-per-WMMA (32 for bf16
// vs 4 for f32 WMMA). GEMM k-loop is double-buffered so global_load_b128
// for step k+1 issue before the WMMAs of step k (non-zero loadcnt waits).
// ---------------------------------------------------------------------------

typedef __bf16 bf16_t;
typedef __attribute__((ext_vector_type(16))) bf16_t v16bf;
typedef __attribute__((ext_vector_type(8)))  bf16_t v8bf;   // 16B -> global_load_b128
typedef __attribute__((ext_vector_type(8)))  float  v8f;

__device__ __forceinline__ v16bf load_frag(const bf16_t* p0, const bf16_t* p1) {
  union { v16bf v; v8bf h[2]; } u;
  u.h[0] = *(const v8bf*)p0;
  u.h[1] = *(const v8bf*)p1;
  return u.v;
}

__device__ __forceinline__ v8f wmma_bf16(v16bf a, v16bf b, v8f c) {
  return __builtin_amdgcn_wmma_f32_16x16x32_bf16(false, a, false, b, (short)0, c,
                                                 false, false);
}

// wave-private LDS reuse: hw keeps DS in-order per wave; fence stops compiler
// reordering and drains DScnt.
#define LDS_FENCE() asm volatile("s_wait_dscnt 0x0" ::: "memory")

// --------------------------- weight transpose ------------------------------
// src: f32 [batch][R][C] -> dst: bf16 [batch][C][R]
__global__ void k_transpose_bf16(const float* __restrict__ src,
                                 bf16_t* __restrict__ dst, int R, int C) {
  size_t slice = (size_t)R * C;
  src += (size_t)blockIdx.y * slice;
  dst += (size_t)blockIdx.y * slice;
  int idx = blockIdx.x * blockDim.x + threadIdx.x;
  if (idx < R * C) {
    int r = idx / C, c = idx % C;
    dst[(size_t)c * R + r] = (bf16_t)src[idx];
  }
}

// ------------------------------- embedding ---------------------------------
__global__ void k_embed(const int* __restrict__ ids, const float* __restrict__ emb,
                        float* __restrict__ xf, bf16_t* __restrict__ xs0,
                        int n, int S, int D) {
  int row = blockIdx.x;              // 0 .. B*n-1
  int b = row / n, pos = row % n;
  int id = (pos == 0) ? 0 : ids[b * S + pos - 1];
  const float* e = emb + (size_t)id * D;
  for (int i = threadIdx.x; i < D; i += blockDim.x) {
    float v = e[i];
    xf[(size_t)row * D + i] = v;
    xs0[(size_t)row * D + i] = (bf16_t)v;
  }
}

// ------------------------------ bf16 GEMM ----------------------------------
// C[M,N] = A[M,K] * Bt[N,K]^T ; optional f32 residual/out, bf16 out,
// transposed bf16 out (CbT[b][col][pos], pos = row % tRows, ld = tLd).
// Wave tile: 16 rows x 64 cols, software-pipelined over K.
__global__ void k_gemm_bf16(const bf16_t* __restrict__ A,
                            const bf16_t* __restrict__ Bt,
                            const float* __restrict__ Res,
                            float* __restrict__ Cf,
                            bf16_t* __restrict__ Cb,
                            bf16_t* __restrict__ CbT,
                            int M, int N, int K, int tRows, int tLd) {
  const int lane = threadIdx.x & 31;
  const int wid  = blockIdx.x * (blockDim.x >> 5) + (threadIdx.x >> 5);
  const int ntN  = N >> 6;
  const int mt = wid / ntN, nt = wid % ntN;
  if (mt * 16 >= M) return;
  const int l16 = lane & 15, hw = lane >> 4;
  int arow = mt * 16 + l16; if (arow >= M) arow = M - 1;   // clamp; store guarded
  const bf16_t* ap = A + (size_t)arow * K;
  const bf16_t* bp[4];
#pragma unroll
  for (int j = 0; j < 4; ++j)
    bp[j] = Bt + (size_t)(nt * 64 + j * 16 + l16) * K + 16 * hw;

  v8f acc[4] = {};
  // prologue: fragments for kk = 0
  v16bf af = load_frag(ap + 16 * hw, ap + 16 * hw + 8);
  v16bf bfr[4];
#pragma unroll
  for (int j = 0; j < 4; ++j) bfr[j] = load_frag(bp[j], bp[j] + 8);

  for (int kk = 0; kk < K; kk += 32) {
    int kn = kk + 32; if (kn >= K) kn = 0;   // wrapped prefetch (discarded on exit)
    // issue next step's 10 x global_load_b128 before consuming current frags
    v16bf afn = load_frag(ap + kn + 16 * hw, ap + kn + 16 * hw + 8);
    v16bf bfn[4];
#pragma unroll
    for (int j = 0; j < 4; ++j) bfn[j] = load_frag(bp[j] + kn, bp[j] + kn + 8);
#pragma unroll
    for (int j = 0; j < 4; ++j) acc[j] = wmma_bf16(af, bfr[j], acc[j]);
    af = afn;
#pragma unroll
    for (int j = 0; j < 4; ++j) bfr[j] = bfn[j];
  }
#pragma unroll
  for (int j = 0; j < 4; ++j) {
#pragma unroll
    for (int g = 0; g < 8; ++g) {
      int row = mt * 16 + g + 8 * hw;
      if (row < M) {
        int col = nt * 64 + j * 16 + l16;
        size_t idx = (size_t)row * N + col;
        float v = acc[j][g];
        if (Res) v += Res[idx];
        if (Cf)  Cf[idx] = v;
        if (Cb)  Cb[idx] = (bf16_t)v;
        if (CbT) {
          int bb = row / tRows, pos = row % tRows;
          CbT[((size_t)bb * N + col) * tLd + pos] = (bf16_t)v;
        }
      }
    }
  }
}

// --------------------------- flash attention -------------------------------
// One wave per (batch, head, 16-query tile); streams 32-key chunks.
// Q,K: bf16 [B*n, H*64] row-major. Vt: bf16 [B][H*64][vtLd] (transposed).
// O:  bf16 [B*n, H*64].
__global__ void k_attn(const bf16_t* __restrict__ Q, const bf16_t* __restrict__ Kc,
                       const bf16_t* __restrict__ Vt, bf16_t* __restrict__ O,
                       int n, int H, int Bc, int vtLd) {
  __shared__ bf16_t plds[8][16][32];
  const int lane = threadIdx.x & 31;
  const int wl   = threadIdx.x >> 5;
  const int wid  = blockIdx.x * (blockDim.x >> 5) + wl;
  const int qtiles = (n + 15) >> 4;
  const int bh = wid / qtiles, qt = wid % qtiles;
  if (bh >= Bc * H) return;
  const int b = bh / H, h = bh % H;
  const int D = H * 64;
  const int l16 = lane & 15, hw = lane >> 4;
  const size_t base = ((size_t)b * n) * D + h * 64;
  const int qbase = qt * 16;

  int qr = qbase + l16; if (qr >= n) qr = n - 1;
  const bf16_t* qp = Q + base + (size_t)qr * D;
  v16bf qf0 = load_frag(qp + 16 * hw, qp + 16 * hw + 8);         // dk 0..31
  v16bf qf1 = load_frag(qp + 32 + 16 * hw, qp + 32 + 16 * hw + 8); // dk 32..63

  float mrow[8], lrow[8];
  v8f oacc[4] = {};
#pragma unroll
  for (int g = 0; g < 8; ++g) { mrow[g] = -1e30f; lrow[g] = 0.f; }

  const int qmax = (qbase + 15 < n) ? qbase + 15 : n - 1;
  const int nch = (qmax >> 5) + 1;

  for (int ch = 0; ch < nch; ++ch) {
    const int kbase = ch * 32;
    v8f s0 = {}, s1 = {};
    {
      int k0 = kbase + l16;      if (k0 >= n) k0 = n - 1;
      int k1 = kbase + 16 + l16; if (k1 >= n) k1 = n - 1;
      const bf16_t* kp0 = Kc + base + (size_t)k0 * D;
      const bf16_t* kp1 = Kc + base + (size_t)k1 * D;
      s0 = wmma_bf16(qf0, load_frag(kp0 + 16 * hw, kp0 + 16 * hw + 8), s0);
      s0 = wmma_bf16(qf1, load_frag(kp0 + 32 + 16 * hw, kp0 + 32 + 16 * hw + 8), s0);
      s1 = wmma_bf16(qf0, load_frag(kp1 + 16 * hw, kp1 + 16 * hw + 8), s1);
      s1 = wmma_bf16(qf1, load_frag(kp1 + 32 + 16 * hw, kp1 + 32 + 16 * hw + 8), s1);
    }
    // scale + causal mask + online softmax (rows live across 16-lane halves)
#pragma unroll
    for (int g = 0; g < 8; ++g) {
      const int q = qbase + g + 8 * hw;
      const int key0 = kbase + l16, key1 = kbase + 16 + l16;
      float v0 = s0[g] * 0.125f;
      float v1 = s1[g] * 0.125f;
      if (key0 > q || key0 >= n || q >= n) v0 = -1e30f;
      if (key1 > q || key1 >= n || q >= n) v1 = -1e30f;
      float mloc = fmaxf(v0, v1);
#pragma unroll
      for (int off = 1; off < 16; off <<= 1)
        mloc = fmaxf(mloc, __shfl_xor(mloc, off));
      float mn = fmaxf(mrow[g], mloc);
      float corr = __expf(mrow[g] - mn);
      mrow[g] = mn;
      float p0 = __expf(v0 - mn), p1 = __expf(v1 - mn);
      float ls = p0 + p1;
#pragma unroll
      for (int off = 1; off < 16; off <<= 1) ls += __shfl_xor(ls, off);
      lrow[g] = lrow[g] * corr + ls;
      oacc[0][g] *= corr; oacc[1][g] *= corr;
      oacc[2][g] *= corr; oacc[3][g] *= corr;
      s0[g] = p0; s1[g] = p1;
    }
    // C-layout -> A-layout re-stage of P through wave-private LDS
#pragma unroll
    for (int g = 0; g < 8; ++g) {
      plds[wl][g + 8 * hw][l16]      = (bf16_t)s0[g];
      plds[wl][g + 8 * hw][16 + l16] = (bf16_t)s1[g];
    }
    LDS_FENCE();
    const bf16_t* pr = &plds[wl][l16][0];
    v16bf pf = load_frag(pr + 16 * hw, pr + 16 * hw + 8);
    LDS_FENCE();
    // O += P @ V  (V transposed: contiguous along keys)
#pragma unroll
    for (int j = 0; j < 4; ++j) {
      const bf16_t* vp =
          Vt + ((size_t)b * D + h * 64 + j * 16 + l16) * vtLd + kbase + 16 * hw;
      v16bf vf = load_frag(vp, vp + 8);
      oacc[j] = wmma_bf16(pf, vf, oacc[j]);
    }
  }
#pragma unroll
  for (int g = 0; g < 8; ++g) {
    const int q = qbase + g + 8 * hw;
    float inv = lrow[g] > 0.f ? 1.f / lrow[g] : 0.f;
    if (q < n) {
#pragma unroll
      for (int j = 0; j < 4; ++j)
        O[base + (size_t)q * D + j * 16 + l16] = (bf16_t)(oacc[j][g] * inv);
    }
  }
}

// ----------------------- logits + vocab softmax ----------------------------
// Wave computes full 16x256 tile (K=1024), then row softmax -> f32 out.
__global__ void k_out_softmax(const bf16_t* __restrict__ Xs,
                              const bf16_t* __restrict__ Wt,
                              float* __restrict__ out, int Mtot) {
  const int K = 1024, V = 256;
  const int lane = threadIdx.x & 31;
  const int wid  = blockIdx.x * (blockDim.x >> 5) + (threadIdx.x >> 5);
  if (wid * 16 >= Mtot) return;
  const int l16 = lane & 15, hw = lane >> 4;
  int ar = wid * 16 + l16; if (ar >= Mtot) ar = Mtot - 1;
  const bf16_t* ap = Xs + (size_t)ar * K;

  v8f acc[16] = {};
  for (int kk = 0; kk < K; kk += 32) {
    v16bf af = load_frag(ap + kk + 16 * hw, ap + kk + 16 * hw + 8);
#pragma unroll
    for (int j = 0; j < 16; ++j) {
      const bf16_t* bp = Wt + (size_t)(j * 16 + l16) * K + kk + 16 * hw;
      v16bf bfr = load_frag(bp, bp + 8);
      acc[j] = wmma_bf16(af, bfr, acc[j]);
    }
  }
#pragma unroll
  for (int g = 0; g < 8; ++g) {
    int row = wid * 16 + g + 8 * hw;
    float mloc = -3.0e38f;
#pragma unroll
    for (int j = 0; j < 16; ++j) mloc = fmaxf(mloc, acc[j][g]);
#pragma unroll
    for (int off = 1; off < 16; off <<= 1)
      mloc = fmaxf(mloc, __shfl_xor(mloc, off));
    float s = 0.f, e[16];
#pragma unroll
    for (int j = 0; j < 16; ++j) { e[j] = __expf(acc[j][g] - mloc); s += e[j]; }
#pragma unroll
    for (int off = 1; off < 16; off <<= 1) s += __shfl_xor(s, off);
    float inv = 1.f / s;
    if (row < Mtot) {
#pragma unroll
      for (int j = 0; j < 16; ++j)
        out[(size_t)row * V + j * 16 + l16] = e[j] * inv;
    }
  }
}

// ------------------------------- launcher ----------------------------------
extern "C" void kernel_launch(void* const* d_in, const int* in_sizes, int n_in,
                              void* d_out, int out_size, void* d_ws, size_t ws_size,
                              hipStream_t stream) {
  (void)in_sizes; (void)n_in; (void)out_size; (void)ws_size;
  const int*   ids  = (const int*)d_in[0];
  const float* emb  = (const float*)d_in[1];
  const float* Wq   = (const float*)d_in[2];
  const float* Wk   = (const float*)d_in[3];
  const float* Wv   = (const float*)d_in[4];
  const float* Wo   = (const float*)d_in[5];
  const float* Wout = (const float*)d_in[6];
  float* out = (float*)d_out;

  constexpr int B = 2, S = 2048, n = 2049, D = 1024, H = 16, L = 16, Vv = 256;
  constexpr int Bn = B * n;          // 4098
  constexpr int VT_LD = 2080;        // padded key stride for transposed V

  char* ws = (char*)d_ws;
  size_t off = 0;
  auto alloc = [&](size_t bytes) {
    char* p = ws + off;
    off += (bytes + 255) & ~(size_t)255;
    return p;
  };
  bf16_t* xs    = (bf16_t*)alloc((size_t)(L + 1) * Bn * D * 2);  // all layer acts
  float*  xf    = (float*) alloc((size_t)Bn * D * 4);            // f32 residual
  bf16_t* Qb    = (bf16_t*)alloc((size_t)Bn * D * 2);
  bf16_t* Kb    = (bf16_t*)alloc((size_t)Bn * D * 2);
  bf16_t* VtB   = (bf16_t*)alloc((size_t)B * D * VT_LD * 2);
  bf16_t* Ob    = (bf16_t*)alloc((size_t)Bn * D * 2);
  bf16_t* WqT   = (bf16_t*)alloc((size_t)L * D * D * 2);
  bf16_t* WkT   = (bf16_t*)alloc((size_t)L * D * D * 2);
  bf16_t* WvT   = (bf16_t*)alloc((size_t)L * D * D * 2);
  bf16_t* WoT   = (bf16_t*)alloc((size_t)L * D * D * 2);
  bf16_t* WoutT = (bf16_t*)alloc((size_t)Vv * D * 2);

  hipMemsetAsync(VtB, 0, (size_t)B * D * VT_LD * 2, stream);  // NaN-proof pads

  {
    dim3 g((D * D + 255) / 256, L);
    k_transpose_bf16<<<g, 256, 0, stream>>>(Wq, WqT, D, D);
    k_transpose_bf16<<<g, 256, 0, stream>>>(Wk, WkT, D, D);
    k_transpose_bf16<<<g, 256, 0, stream>>>(Wv, WvT, D, D);
    k_transpose_bf16<<<g, 256, 0, stream>>>(Wo, WoT, D, D);
    dim3 g2((D * Vv + 255) / 256, 1);
    k_transpose_bf16<<<g2, 256, 0, stream>>>(Wout, WoutT, D, Vv);
  }
  k_embed<<<Bn, 256, 0, stream>>>(ids, emb, xf, xs, n, S, D);

  const int Mtiles = (Bn + 15) / 16;                 // 257
  const int gemmBlocks = (Mtiles * (D / 64) + 7) / 8;
  const int attnBlocks = (B * H * ((n + 15) / 16) + 7) / 8;

  for (int l = 0; l < L; ++l) {
    const bf16_t* xin = xs + (size_t)l * Bn * D;
    k_gemm_bf16<<<gemmBlocks, 256, 0, stream>>>(
        xin, WqT + (size_t)l * D * D, nullptr, nullptr, Qb, nullptr,
        Bn, D, D, 0, 0);
    k_gemm_bf16<<<gemmBlocks, 256, 0, stream>>>(
        xin, WkT + (size_t)l * D * D, nullptr, nullptr, Kb, nullptr,
        Bn, D, D, 0, 0);
    k_gemm_bf16<<<gemmBlocks, 256, 0, stream>>>(
        xin, WvT + (size_t)l * D * D, nullptr, nullptr, nullptr, VtB,
        Bn, D, D, n, VT_LD);
    k_attn<<<attnBlocks, 256, 0, stream>>>(Qb, Kb, VtB, Ob, n, H, B, VT_LD);
    k_gemm_bf16<<<gemmBlocks, 256, 0, stream>>>(
        Ob, WoT + (size_t)l * D * D, xf, xf, xs + (size_t)(l + 1) * Bn * D,
        nullptr, Bn, D, D, 0, 0);
  }
  const int Mtot = (L + 1) * Bn;                     // 69666
  k_out_softmax<<<(((Mtot + 15) / 16) + 7) / 8, 256, 0, stream>>>(
      xs, WoutT, out, Mtot);
}
